// SigLossTCNStyleCond_44727789420653
// MI455X (gfx1250) — matrix operand order repaired and checked by reading, not compiled, not run on hardware
//
#include <hip/hip_runtime.h>
#include <hip/hip_bf16.h>

#ifndef USE_TDM
#define USE_TDM 1   // Tensor Data Mover staging of weight tiles into LDS
#endif

typedef float v8f  __attribute__((ext_vector_type(8)));
typedef __bf16 v16bf __attribute__((ext_vector_type(16)));
typedef unsigned int v4u __attribute__((ext_vector_type(4)));
typedef int v8i __attribute__((ext_vector_type(8)));
typedef int v4i __attribute__((ext_vector_type(4)));

union Frag { uint4 u[2]; v16bf v; };

// ---------------------------------------------------------------------------
// Input pack: x (B,T,D) fp32 -> act (B,T,CinPad) bf16, zero channel pad
// ---------------------------------------------------------------------------
__global__ __launch_bounds__(256) void k_init_act(const float* __restrict__ x,
                                                  __hip_bfloat16* __restrict__ act,
                                                  int D, int CinPad, long total) {
  long idx = (long)blockIdx.x * blockDim.x + threadIdx.x;
  if (idx >= total) return;
  int c = (int)(idx % CinPad);
  long bt = idx / CinPad;
  float v = (c < D) ? x[bt * D + c] : 0.0f;
  act[idx] = __float2bfloat16(v);
}

// ---------------------------------------------------------------------------
// Weight prep: w (Cout,Cin,Kw) fp32 -> bf16 B-fragment order.
//   lane n   (0..15): column co = ct*16+n, K = kc*32 + 0..7, 16..23
//   lane 16+n       : column co = ct*16+n, K = kc*32 + 8..15, 24..31
// stored: frag[(k*KC+kc)*NT+ct][lane*16 + i]
// ---------------------------------------------------------------------------
__global__ __launch_bounds__(256) void k_prep_w(const float* __restrict__ w,
                                                __hip_bfloat16* __restrict__ wfrag,
                                                int Kw, int Cin, int CinPad,
                                                int Cout, int CoutPad) {
  int KC = CinPad >> 5, NT = CoutPad >> 4;
  long total = (long)Kw * KC * NT * 512;
  long tid = (long)blockIdx.x * blockDim.x + threadIdx.x;
  if (tid >= total) return;
  int i    = (int)(tid & 15);
  int lane = (int)((tid >> 4) & 31);
  int fi   = (int)(tid >> 9);
  int ct = fi % NT;
  int kc = (fi / NT) % KC;
  int k  = fi / (NT * KC);
  int n = lane & 15, hi = lane >> 4;
  int co = ct * 16 + n;
  int ci = kc * 32 + (i < 8 ? hi * 8 + i : 16 + hi * 8 + (i - 8));
  float v = 0.0f;
  if (co < Cout && ci < Cin) v = w[((long)co * Cin + ci) * Kw + k];
  wfrag[tid] = __float2bfloat16(v);
}

// ---------------------------------------------------------------------------
// Implicit-GEMM dilated conv1d, bf16 WMMA, f32 accum, register-blocked.
// Block: 64 rows x 64 cols. Wave w: rows [blk*64 + w*16, +16), all 4 N-tiles.
// Weights for the block's 64-col slice staged into LDS (TDM async copy).
// act_in (B,Lin,CinPad) bf16 -> act_out (B,Lout,CoutPad) bf16.
// ---------------------------------------------------------------------------
__global__ __launch_bounds__(128) void k_conv_wmma(
    const __hip_bfloat16* __restrict__ act_in,
    const __hip_bfloat16* __restrict__ wfrag,
    const float* __restrict__ bias,
    __hip_bfloat16* __restrict__ act_out,
    int Lin, int Lout, int CinPad, int Cout, int CoutPad,
    int Kw, int dil, int pad, int do_relu) {
  extern __shared__ __hip_bfloat16 smem[];   // nslab * 2048 bf16 (nslab*4KB)
  const int KC = CinPad >> 5;
  const int NT = CoutPad >> 4;
  const int nslab = Kw * KC;                 // K-slabs of 32
  const int wave = threadIdx.x >> 5;
  const int lane = threadIdx.x & 31;
  const int laneM = lane & 15;
  const int hi = lane >> 4;
  const int ct0 = blockIdx.y * 4;            // first of this block's 4 N tiles
  const int m_base = blockIdx.x * 64 + wave * 16;

  // ---- stage weight fragments: nslab rows x 4KB, global row stride NT*1KB --
#if USE_TDM
  if (wave == 0) {
    const unsigned lds_addr = (unsigned)(uintptr_t)(&smem[0]);
    const unsigned long long gaddr =
        (unsigned long long)(uintptr_t)(wfrag + (size_t)ct0 * 512);
    const unsigned rows = (unsigned)nslab;
    const unsigned stride_dw = (unsigned)NT * 256u;   // dwords between slabs
    // D# group0: count=1 | lds_addr | global_addr(57b) | type=2
    v4u g0 = { 1u, lds_addr,
               (unsigned)(gaddr & 0xffffffffu),
               (unsigned)(((gaddr >> 32) & 0x01ffffffu) | (2u << 30)) };
    // D# group1: data_size=4B; tensor_dim0=tile_dim0=1024 dw; dim1=tile_dim1=rows
    v8i g1 = { (int)(2u << 16),                 // data_size=2 (4B), mask=0
               (int)(1024u << 16),              // tensor_dim0[15:0]
               (int)((rows & 0xffffu) << 16),   // dim0[31:16]=0 | tensor_dim1[15:0]
               (int)(((rows >> 16) & 0xffffu) | (1024u << 16)), // dim1[31:16] | tile_dim0
               (int)rows,                       // tile_dim1 | tile_dim2=0
               (int)stride_dw,                  // tensor_dim0_stride[31:0]
               0, 0 };
    v4i gz4 = { 0, 0, 0, 0 };
    v8i gz8 = { 0, 0, 0, 0, 0, 0, 0, 0 };
    __builtin_amdgcn_tensor_load_to_lds(g0, g1, gz4, gz4, gz8, 0);
    __builtin_amdgcn_s_wait_tensorcnt((short)0);
  }
  __syncthreads();
#else
  {
    const __hip_bfloat16* src = wfrag + (size_t)ct0 * 512;
    const int chunks = nslab * 256;            // 16B chunks
    for (int c = threadIdx.x; c < chunks; c += 128) {
      int slab = c >> 8, off = c & 255;
      *(uint4*)(smem + (size_t)slab * 2048 + off * 8) =
          *(const uint4*)(src + (size_t)slab * NT * 512 + off * 8);
    }
    __syncthreads();
  }
#endif

  const int r = m_base + laneM;
  const int b = r / Lout;
  const int t = r - b * Lout;
  const long rowB = (long)b * Lin * CinPad;

  const v8f z8 = {0.f, 0.f, 0.f, 0.f, 0.f, 0.f, 0.f, 0.f};
  v8f acc[4];
  acc[0] = z8; acc[1] = z8; acc[2] = z8; acc[3] = z8;

  for (int k = 0; k < Kw; ++k) {
    const int tsrc = t + k * dil - pad;
    const bool valid = (tsrc >= 0) && (tsrc < Lin);
    const __hip_bfloat16* arow =
        act_in + rowB + (long)(valid ? tsrc : 0) * CinPad + hi * 8;
    const __hip_bfloat16* lw = smem + (long)(k * KC) * 2048 + lane * 16;
    for (int kc = 0; kc < KC; ++kc) {
      Frag a;
      if (valid) {
        a.u[0] = *(const uint4*)(arow);        // K = hi*8 .. hi*8+7
        a.u[1] = *(const uint4*)(arow + 16);   // K = 16+hi*8 ..
      } else {
        a.u[0] = make_uint4(0u, 0u, 0u, 0u);
        a.u[1] = make_uint4(0u, 0u, 0u, 0u);
      }
      __builtin_prefetch(arow + 32, 0, 1);
#pragma unroll
      for (int n = 0; n < 4; ++n) {
        Frag bb;
        bb.u[0] = *(const uint4*)(lw + n * 512);
        bb.u[1] = *(const uint4*)(lw + n * 512 + 8);
        acc[n] = __builtin_amdgcn_wmma_f32_16x16x32_bf16(
            false, a.v, false, bb.v, (short)0, acc[n], false, false);
      }
      arow += 32;
      lw += 2048;
    }
  }

  // C/D layout: lane n<16 -> col n rows 0..7; lane 16+n -> col n rows 8..15
  const int n15 = lane & 15;
#pragma unroll
  for (int n = 0; n < 4; ++n) {
    const int co = (ct0 + n) * 16 + n15;
    const float bv = (co < Cout) ? bias[co] : 0.0f;
#pragma unroll
    for (int i = 0; i < 8; ++i) {
      const int rr = m_base + hi * 8 + i;
      const int ob = rr / Lout;
      const int ot = rr - ob * Lout;
      float v = acc[n][i] + bv;
      if (do_relu) v = fmaxf(v, 0.0f);
      if (co >= Cout) v = 0.0f;                // keep channel pad zeroed
      act_out[((long)ob * Lout + ot) * CoutPad + co] = __float2bfloat16(v);
    }
  }
}

// ---------------------------------------------------------------------------
// LayerNorm over channels at each (b,t): one wave32 per row.
// ---------------------------------------------------------------------------
__global__ __launch_bounds__(32) void k_layernorm(__hip_bfloat16* __restrict__ act,
                                                  const float* __restrict__ g,
                                                  const float* __restrict__ bt,
                                                  int Cout, int CoutPad) {
  __hip_bfloat16* row = act + (long)blockIdx.x * CoutPad;
  const int lane = threadIdx.x;
  float s = 0.0f, s2 = 0.0f;
  for (int c = lane; c < Cout; c += 32) {
    float v = __bfloat162float(row[c]);
    s += v; s2 += v * v;
  }
#pragma unroll
  for (int off = 16; off > 0; off >>= 1) {
    s  += __shfl_xor(s, off, 32);
    s2 += __shfl_xor(s2, off, 32);
  }
  const float mean = s / Cout;
  const float var = s2 / Cout - mean * mean;
  const float rstd = rsqrtf(var + 1e-5f);
  for (int c = lane; c < Cout; c += 32) {
    float v = __bfloat162float(row[c]);
    v = (v - mean) * rstd * g[c] + bt[c];
    row[c] = __float2bfloat16(v);
  }
}

// z[b,co] = mean_t h[b,t,co]  -> bf16 into head input cols [0,256)
__global__ __launch_bounds__(256) void k_zmean(const __hip_bfloat16* __restrict__ act,
                                               __hip_bfloat16* __restrict__ headin,
                                               int Lout, int CoutPad) {
  const int b = blockIdx.x, co = threadIdx.x;
  const __hip_bfloat16* p = act + (long)b * Lout * CoutPad + co;
  float s = 0.0f;
  for (int t = 0; t < Lout; ++t) s += __bfloat162float(p[(long)t * CoutPad]);
  headin[(long)b * 288 + co] = __float2bfloat16(s / (float)Lout);
}

// ---------------------------------------------------------------------------
// Style statistics feats[b, 5*D] (fp32, matches reference clip/ddof semantics)
// ---------------------------------------------------------------------------
__global__ __launch_bounds__(128) void k_feats(const float* __restrict__ x,
                                               float* __restrict__ feats,
                                               int B, int T, int D) {
  const int b = blockIdx.x * blockDim.x + threadIdx.x;
  if (b >= B) return;
  const float* xb = x + (long)b * T * D;
  float* f = feats + (long)b * 5 * D;
  for (int d = 0; d < 8; ++d) {
    float s = 0.f, s2 = 0.f;
    for (int t = 0; t < T; ++t) { float v = xb[t * D + d]; s += v; s2 += v * v; }
    const float mu = s / T;
    float varn = (s2 - T * mu * mu) / (T - 1);               // ddof=1
    float sd = sqrtf(fmaxf(varn, 0.f)); sd = fmaxf(sd, 1e-6f);
    const float delta = xb[(T - 1) * D + d] - xb[d];
    float sxy = 0.f, sx0 = 0.f, sx1 = 0.f;
    for (int t = 0; t < T - 1; ++t) {
      float a = xb[t * D + d], c = xb[(t + 1) * D + d];
      sxy += a * c; sx0 += a * a; sx1 += c * c;
    }
    const float num = sxy / (T - 1);
    float den = sqrtf((sx0 / (T - 1)) * (sx1 / (T - 1)));
    den = fmaxf(den, 1e-6f);
    f[d] = mu; f[8 + d] = sd; f[16 + d] = delta; f[24 + d] = num / den;
  }
  float rs = 0.f, rs2 = 0.f;
  for (int t = 0; t < T - 1; ++t) {
    float r = xb[(t + 1) * D] - xb[t * D];
    rs += r; rs2 += r * r;
  }
  const float rmu = rs / (T - 1);
  const float rvol = sqrtf(fmaxf((rs2 - (T - 1) * rmu * rmu) / (T - 2), 0.f)); // ddof=1
  for (int d = 0; d < 8; ++d) f[32 + d] = rvol;
}

__global__ __launch_bounds__(128) void k_style1(const float* __restrict__ feats,
                                                const float* __restrict__ w1,
                                                const float* __restrict__ b1,
                                                float* __restrict__ sh, int B) {
  const int idx = blockIdx.x * blockDim.x + threadIdx.x;
  if (idx >= B * 60) return;
  const int b = idx / 60, j = idx - b * 60;
  const float* f = feats + (long)b * 40;
  const float* w = w1 + (long)j * 40;
  float s = b1[j];
  for (int k = 0; k < 40; ++k) s += f[k] * w[k];
  sh[idx] = fmaxf(s, 0.0f);
}

__global__ __launch_bounds__(128) void k_style2(const float* __restrict__ sh,
                                                const float* __restrict__ w2,
                                                const float* __restrict__ b2,
                                                __hip_bfloat16* __restrict__ headin,
                                                int B) {
  const int idx = blockIdx.x * blockDim.x + threadIdx.x;
  if (idx >= B * 32) return;
  const int b = idx / 32, j = idx - b * 32;
  const float* h = sh + (long)b * 60;
  const float* w = w2 + (long)j * 60;
  float s = b2[j];
  for (int k = 0; k < 60; ++k) s += h[k] * w[k];
  headin[(long)b * 288 + 256 + j] = __float2bfloat16(s);
}

// head layer2 (256->10) + cumsum -> y_pred_levels & r_pred
__global__ __launch_bounds__(128) void k_head2(const __hip_bfloat16* __restrict__ hid,
                                               const float* __restrict__ w2,
                                               const float* __restrict__ b2,
                                               const float* __restrict__ p0,
                                               float* __restrict__ y_out,
                                               float* __restrict__ r_out, int B) {
  const int b = blockIdx.x * blockDim.x + threadIdx.x;
  if (b >= B) return;
  const __hip_bfloat16* h = hid + (long)b * 256;
  float r[10];
#pragma unroll
  for (int j = 0; j < 10; ++j) {
    const float* w = w2 + (long)j * 256;
    float s = b2[j];
    for (int c = 0; c < 256; ++c) s += __bfloat162float(h[c]) * w[c];
    r[j] = s;
  }
  float cum = p0[b];
#pragma unroll
  for (int j = 0; j < 10; ++j) {
    cum += r[j];
    y_out[b * 10 + j] = cum;
    r_out[b * 10 + j] = r[j];
  }
}

// ---------------------------------------------------------------------------
// logsig3(augment(levels)): levels (B,10), augmented d=4, out (B, 4+16+64=84)
// ---------------------------------------------------------------------------
__global__ __launch_bounds__(128) void k_logsig(const float* __restrict__ levels,
                                                float* __restrict__ out, int B) {
  const int b = blockIdx.x * blockDim.x + threadIdx.x;
  if (b >= B) return;
  float ell[10];
#pragma unroll
  for (int t = 0; t < 10; ++t) ell[t] = levels[b * 10 + t];

  float prev[4] = {0.f, 0.f, 0.f, 0.f};
  float S1[4] = {0.f, 0.f, 0.f, 0.f};
  float S2[16]; float S3[64];
#pragma unroll
  for (int i = 0; i < 16; ++i) S2[i] = 0.f;
#pragma unroll
  for (int i = 0; i < 64; ++i) S3[i] = 0.f;

  const float inv9 = 1.0f / 9.0f;
#pragma unroll
  for (int j = 0; j < 10; ++j) {
    const float tj = (float)j * inv9;
    const float lagl = (j == 0) ? ell[0] : ell[j - 1];
    const float lagt = (j == 0) ? 0.0f : (float)(j - 1) * inv9;
    float cur[4] = {ell[j], tj, lagl, lagt};
    float v[4];
#pragma unroll
    for (int i = 0; i < 4; ++i) { v[i] = cur[i] - prev[i]; prev[i] = cur[i]; }
    float v2[16];
#pragma unroll
    for (int i = 0; i < 4; ++i)
#pragma unroll
      for (int k = 0; k < 4; ++k) v2[i * 4 + k] = v[i] * v[k];
#pragma unroll
    for (int i = 0; i < 4; ++i)
#pragma unroll
      for (int jj = 0; jj < 4; ++jj)
#pragma unroll
        for (int k = 0; k < 4; ++k)
          S3[i * 16 + jj * 4 + k] += S2[i * 4 + jj] * v[k]
                                   + 0.5f * S1[i] * v2[jj * 4 + k]
                                   + v2[i * 4 + jj] * v[k] * (1.0f / 6.0f);
#pragma unroll
    for (int i = 0; i < 4; ++i)
#pragma unroll
      for (int jj = 0; jj < 4; ++jj)
        S2[i * 4 + jj] += S1[i] * v[jj] + 0.5f * v2[i * 4 + jj];
#pragma unroll
    for (int i = 0; i < 4; ++i) S1[i] += v[i];
  }

  float* o = out + (long)b * 84;
#pragma unroll
  for (int i = 0; i < 4; ++i) o[i] = S1[i];
#pragma unroll
  for (int i = 0; i < 4; ++i)
#pragma unroll
    for (int jj = 0; jj < 4; ++jj)
      o[4 + i * 4 + jj] = S2[i * 4 + jj] - 0.5f * S1[i] * S1[jj];
#pragma unroll
  for (int i = 0; i < 4; ++i)
#pragma unroll
    for (int jj = 0; jj < 4; ++jj)
#pragma unroll
      for (int k = 0; k < 4; ++k)
        o[20 + i * 16 + jj * 4 + k] =
            S3[i * 16 + jj * 4 + k]
            - 0.5f * (S1[i] * S2[jj * 4 + k] + S2[i * 4 + jj] * S1[k])
            + S1[i] * S1[jj] * S1[k] * (1.0f / 3.0f);
}

// ---------------------------------------------------------------------------
extern "C" void kernel_launch(void* const* d_in, const int* in_sizes, int n_in,
                              void* d_out, int out_size, void* d_ws, size_t ws_size,
                              hipStream_t stream) {
  (void)in_sizes; (void)n_in; (void)out_size; (void)ws_size;
  const int B = 2048, T = 60, D = 8, STEPS = 10;

  const float* x        = (const float*)d_in[0];
  const float* y_true   = (const float*)d_in[1];
  const float* p0       = (const float*)d_in[2];
  const float* style_w1 = (const float*)d_in[33];
  const float* style_b1 = (const float*)d_in[34];
  const float* style_w2 = (const float*)d_in[35];
  const float* style_b2 = (const float*)d_in[36];
  const float* head_w1  = (const float*)d_in[37];
  const float* head_b1  = (const float*)d_in[38];
  const float* head_w2  = (const float*)d_in[39];
  const float* head_b2  = (const float*)d_in[40];

  // workspace carve-out (256B aligned)
  char* ws = (char*)d_ws;
  size_t off = 0;
  auto carve = [&](size_t bytes) {
    size_t o = off; off += (bytes + 255) & ~(size_t)255; return (void*)(ws + o);
  };
  __hip_bfloat16* actA    = (__hip_bfloat16*)carve((size_t)B * 192 * 256 * 2);
  __hip_bfloat16* actB    = (__hip_bfloat16*)carve((size_t)B * 192 * 256 * 2);
  __hip_bfloat16* wfrag   = (__hip_bfloat16*)carve((size_t)3 * 288 * 256 * 2);
  __hip_bfloat16* headin  = (__hip_bfloat16*)carve((size_t)B * 288 * 2);
  __hip_bfloat16* headhid = (__hip_bfloat16*)carve((size_t)B * 256 * 2);
  float* feats = (float*)carve((size_t)B * 40 * 4);
  float* sh    = (float*)carve((size_t)B * 60 * 4);

  float* y_out  = (float*)d_out;                 // (B,10,1)
  float* r_out  = y_out + (size_t)B * STEPS;     // (B,10)
  float* S_pred = r_out + (size_t)B * STEPS;     // (B,84)
  float* S_true = S_pred + (size_t)B * 84;       // (B,84)

  // 1) pack input into bf16 (B,T,32) with zeroed channel pad
  {
    long total = (long)B * T * 32;
    k_init_act<<<dim3((unsigned)((total + 255) / 256)), 256, 0, stream>>>(
        x, actA, D, 32, total);
  }

  // 2) TCN stack
  const int couts[5]    = {60, 128, 128, 256, 256};
  const int coutPads[5] = {64, 128, 128, 256, 256};
  int Cin = D, CinPad = 32;
  int L = T;
  for (int i = 0; i < 5; ++i) {
    const int dil = 1 << i, pad = 2 * dil;
    const int Cout = couts[i], CoutPad = coutPads[i];
    const float* w1 = (const float*)d_in[3 + 6 * i + 0];
    const float* b1 = (const float*)d_in[3 + 6 * i + 1];
    const float* w2 = (const float*)d_in[3 + 6 * i + 2];
    const float* b2 = (const float*)d_in[3 + 6 * i + 3];
    const float* g  = (const float*)d_in[3 + 6 * i + 4];
    const float* bt = (const float*)d_in[3 + 6 * i + 5];

    { // conv1: actA (L, CinPad) -> actB (L+pad, CoutPad)
      long total = (long)3 * (CinPad / 32) * (CoutPad / 16) * 512;
      k_prep_w<<<dim3((unsigned)((total + 255) / 256)), 256, 0, stream>>>(
          w1, wfrag, 3, Cin, CinPad, Cout, CoutPad);
      const int Lout = L + pad;
      const unsigned smem_bytes = (unsigned)(3 * (CinPad / 32) * 4096);
      dim3 grid((unsigned)((B * Lout) / 64), (unsigned)(CoutPad / 64));
      k_conv_wmma<<<grid, 128, smem_bytes, stream>>>(actA, wfrag, b1, actB,
                                                     L, Lout, CinPad, Cout, CoutPad,
                                                     3, dil, pad, 1);
      L = Lout;
    }
    { // conv2: actB (L, CoutPad) -> actA (L+pad, CoutPad)
      long total = (long)3 * (CoutPad / 32) * (CoutPad / 16) * 512;
      k_prep_w<<<dim3((unsigned)((total + 255) / 256)), 256, 0, stream>>>(
          w2, wfrag, 3, Cout, CoutPad, Cout, CoutPad);
      const int Lout = L + pad;
      const unsigned smem_bytes = (unsigned)(3 * (CoutPad / 32) * 4096);
      dim3 grid((unsigned)((B * Lout) / 64), (unsigned)(CoutPad / 64));
      k_conv_wmma<<<grid, 128, smem_bytes, stream>>>(actB, wfrag, b2, actA,
                                                     L, Lout, CoutPad, Cout, CoutPad,
                                                     3, dil, pad, 1);
      L = Lout;
    }
    // LayerNorm over channels, in place on actA
    k_layernorm<<<dim3((unsigned)(B * L)), 32, 0, stream>>>(actA, g, bt, Cout, CoutPad);
    Cin = Cout; CinPad = CoutPad;
  }

  // 3) z = mean over time -> head input cols [0,256)
  k_zmean<<<dim3((unsigned)B), 256, 0, stream>>>(actA, headin, L, 256);

  // 4) style branch -> head input cols [256,288)
  k_feats<<<dim3((unsigned)((B + 127) / 128)), 128, 0, stream>>>(x, feats, B, T, D);
  k_style1<<<dim3((unsigned)((B * 60 + 127) / 128)), 128, 0, stream>>>(
      feats, style_w1, style_b1, sh, B);
  k_style2<<<dim3((unsigned)((B * 32 + 127) / 128)), 128, 0, stream>>>(
      sh, style_w2, style_b2, headin, B);

  // 5) head layer1 (288 -> 256) as WMMA GEMM (Kw=1, L=1)
  {
    long total = (long)1 * (288 / 32) * (256 / 16) * 512;
    k_prep_w<<<dim3((unsigned)((total + 255) / 256)), 256, 0, stream>>>(
        head_w1, wfrag, 1, 288, 288, 256, 256);
    const unsigned smem_bytes = (unsigned)(1 * (288 / 32) * 4096);
    dim3 grid((unsigned)(B / 64), (unsigned)(256 / 64));
    k_conv_wmma<<<grid, 128, smem_bytes, stream>>>(headin, wfrag, head_b1, headhid,
                                                   1, 1, 288, 256, 256, 1, 1, 0, 1);
  }

  // 6) head layer2 + cumsum -> y_pred_levels, r_pred
  k_head2<<<dim3((unsigned)((B + 127) / 128)), 128, 0, stream>>>(
      headhid, head_w2, head_b2, p0, y_out, r_out, B);

  // 7) log-signatures of predicted and true paths
  k_logsig<<<dim3((unsigned)((B + 127) / 128)), 128, 0, stream>>>(y_out, S_pred, B);
  k_logsig<<<dim3((unsigned)((B + 127) / 128)), 128, 0, stream>>>(y_true, S_true, B);
}
